// KANLayer_47055661695145
// MI455X (gfx1250) — compile-verified, matrix operand order checked by reading, not run on hardware
//
#include <hip/hip_runtime.h>

typedef _Float16 v8h  __attribute__((ext_vector_type(8)));
typedef _Float16 v16h __attribute__((ext_vector_type(16)));
typedef float    v8f  __attribute__((ext_vector_type(8)));
typedef int      v4i  __attribute__((ext_vector_type(4)));

#define B_DIM 4096
#define I_DIM 1024
#define O_DIM 1024
#define K_RBF 16
#define KK    (I_DIM * K_RBF)   /* 16384 contraction length */

#define BM  128
#define BN  128
#define BK  32
#define LDT 40                  /* padded LDS row stride in f16 elems (32 + 8 pad) */

#if __has_builtin(__builtin_amdgcn_global_load_async_to_lds_b128)
#define USE_ASYNC_LDS 1
#else
#define USE_ASYNC_LDS 0
#endif

__device__ __forceinline__ void wait_async_zero() {
#if __has_builtin(__builtin_amdgcn_s_wait_asynccnt)
  __builtin_amdgcn_s_wait_asynccnt(0);
#else
  asm volatile("s_wait_asynccnt 0x0" ::: "memory");
#endif
}

union FragU { v16h v; v8h h[2]; };

// ---------------------------------------------------------------------------
// Kernel 1: Gaussian RBF basis, fp32 -> f16, act[b, i*16+k]
// ---------------------------------------------------------------------------
__global__ __launch_bounds__(256) void rbf_act_f16_kernel(
    const float* __restrict__ x,
    const float* __restrict__ centers,
    const float* __restrict__ widths,
    _Float16* __restrict__ act)
{
  const int idx = blockIdx.x * 256 + threadIdx.x;   // one per (b, i)
  const float xv = x[idx];
  v8h lo = {}, hi = {};
#pragma unroll
  for (int k = 0; k < K_RBF; ++k) {
    const float c = centers[k];
    const float w = widths[k];
    const float d = xv - c;
    const float e = __expf(-d * d * (0.5f / (w * w)));
    if (k < 8) lo[k]     = (_Float16)e;
    else       hi[k - 8] = (_Float16)e;
  }
  v8h* dst = (v8h*)(act + (size_t)idx * K_RBF);
  dst[0] = lo;
  dst[1] = hi;
}

// ---------------------------------------------------------------------------
// Kernel 2: weights fp32 -> f16 (layout already contraction-major: (O, I*K))
// ---------------------------------------------------------------------------
__global__ __launch_bounds__(256) void cvt_w_f16_kernel(
    const float* __restrict__ w, _Float16* __restrict__ wf)
{
  const int idx = blockIdx.x * 256 + threadIdx.x;   // one per (o, i)
  v8h lo = {}, hi = {};
#pragma unroll
  for (int k = 0; k < K_RBF; ++k) {
    const float v = w[(size_t)idx * K_RBF + k];
    if (k < 8) lo[k]     = (_Float16)v;
    else       hi[k - 8] = (_Float16)v;
  }
  v8h* dst = (v8h*)(wf + (size_t)idx * K_RBF);
  dst[0] = lo;
  dst[1] = hi;
}

// ---------------------------------------------------------------------------
// Kernel 3: C[4096x1024] = act[4096x16384] * W^T, f16 WMMA, f32 accumulate.
// 256 threads = 8 waves; block tile 128x128; wave tile 64x32 (4x2 frags).
// Double-buffered LDS; global->LDS staging via async DMA when available.
// ---------------------------------------------------------------------------
__global__ __launch_bounds__(256) void kan_wmma_gemm_kernel(
    const _Float16* __restrict__ A,   // (B_DIM, KK)
    const _Float16* __restrict__ W,   // (O_DIM, KK)
    const float* __restrict__ bias,   // (O_DIM)
    float* __restrict__ C)            // (B_DIM, O_DIM)
{
  __shared__ _Float16 As[2][BM * LDT];
  __shared__ _Float16 Bs[2][BN * LDT];

  const int tid  = threadIdx.x;
  const int lane = tid & 31;
  const int wave = tid >> 5;
  const int wm   = wave & 1;          // 2 row-slabs of 64
  const int wn   = wave >> 1;         // 4 col-slabs of 32
  const int m0   = blockIdx.y * BM;
  const int n0   = blockIdx.x * BN;

  const int ln16 = lane & 15;
  const int klo  = (lane >> 4) << 3;  // 0 or 8: per-lane K sub-offset (ISA 16-bit frag layout)

  // Staging map: 512 x 16B chunks per matrix tile, 2 per thread.
  const int r0  = tid >> 2;                 // rows 0..63
  const int seg = (tid & 3) * 8;            // f16 elems within row
  const _Float16* gA0 = A + (size_t)(m0 + r0)      * KK + seg;
  const _Float16* gA1 = A + (size_t)(m0 + r0 + 64) * KK + seg;
  const _Float16* gW0 = W + (size_t)(n0 + r0)      * KK + seg;
  const _Float16* gW1 = W + (size_t)(n0 + r0 + 64) * KK + seg;
  const int l0 = r0 * LDT + seg;
  const int l1 = (r0 + 64) * LDT + seg;

  v8f acc[4][2];
#pragma unroll
  for (int mi = 0; mi < 4; ++mi)
#pragma unroll
    for (int ni = 0; ni < 2; ++ni)
      acc[mi][ni] = (v8f){};

#if USE_ASYNC_LDS
  auto stage_async = [&](int b, int k0) {
    __builtin_amdgcn_global_load_async_to_lds_b128((v4i*)(gA0 + k0), (v4i*)&As[b][l0], 0, 0);
    __builtin_amdgcn_global_load_async_to_lds_b128((v4i*)(gA1 + k0), (v4i*)&As[b][l1], 0, 0);
    __builtin_amdgcn_global_load_async_to_lds_b128((v4i*)(gW0 + k0), (v4i*)&Bs[b][l0], 0, 0);
    __builtin_amdgcn_global_load_async_to_lds_b128((v4i*)(gW1 + k0), (v4i*)&Bs[b][l1], 0, 0);
  };
#else
  v8h ra0, ra1, rb0, rb1;
  auto gload = [&](int k0) {
    ra0 = *(const v8h*)(gA0 + k0);
    ra1 = *(const v8h*)(gA1 + k0);
    rb0 = *(const v8h*)(gW0 + k0);
    rb1 = *(const v8h*)(gW1 + k0);
  };
  auto lstore = [&](int b) {
    *(v8h*)&As[b][l0] = ra0;
    *(v8h*)&As[b][l1] = ra1;
    *(v8h*)&Bs[b][l0] = rb0;
    *(v8h*)&Bs[b][l1] = rb1;
  };
#endif

  auto compute = [&](int b) {
    FragU bfr[2], afr[4];
#pragma unroll
    for (int ni = 0; ni < 2; ++ni) {
      const int col = (wn * 32 + ni * 16 + ln16) * LDT;
      bfr[ni].h[0] = *(const v8h*)&Bs[b][col + klo];
      bfr[ni].h[1] = *(const v8h*)&Bs[b][col + klo + 16];
    }
#pragma unroll
    for (int mi = 0; mi < 4; ++mi) {
      const int row = (wm * 64 + mi * 16 + ln16) * LDT;
      afr[mi].h[0] = *(const v8h*)&As[b][row + klo];
      afr[mi].h[1] = *(const v8h*)&As[b][row + klo + 16];
    }
#pragma unroll
    for (int mi = 0; mi < 4; ++mi)
#pragma unroll
      for (int ni = 0; ni < 2; ++ni)
        acc[mi][ni] = __builtin_amdgcn_wmma_f32_16x16x32_f16(
            false, afr[mi].v, false, bfr[ni].v, (short)0, acc[mi][ni], false, false);
  };

#if USE_ASYNC_LDS
  stage_async(0, 0);
  wait_async_zero();
  __syncthreads();
  int buf = 0;
  for (int k0 = 0; k0 < KK; k0 += BK) {
    if (k0 + BK < KK) stage_async(buf ^ 1, k0 + BK);  // DMA next tile, no VGPR round-trip
    compute(buf);                                     // WMMAs overlap the async DMA
    wait_async_zero();
    __syncthreads();
    buf ^= 1;
  }
#else
  gload(0);
  lstore(0);
  __syncthreads();
  int buf = 0;
  for (int k0 = 0; k0 < KK; k0 += BK) {
    if (k0 + BK < KK) gload(k0 + BK);   // issue global loads early...
    compute(buf);                        // ...overlap their latency with WMMAs
    if (k0 + BK < KK) lstore(buf ^ 1);   // loadcnt wait lands here, after compute
    __syncthreads();
    buf ^= 1;
  }
#endif

  // Epilogue: C/D frag layout — VGPR j: lanes0-15 M=j, lanes16-31 M=j+8; N=lane%16.
  const int mb = (lane >> 4) * 8;
#pragma unroll
  for (int ni = 0; ni < 2; ++ni) {
    const int col = n0 + wn * 32 + ni * 16 + ln16;
    const float bv = bias[col];
#pragma unroll
    for (int mi = 0; mi < 4; ++mi) {
#pragma unroll
      for (int j = 0; j < 8; ++j) {
        const int row = m0 + wm * 64 + mi * 16 + mb + j;
        C[(size_t)row * O_DIM + col] = acc[mi][ni][j] + bv;
      }
    }
  }
}

// ---------------------------------------------------------------------------
// Fallback (only if workspace too small): plain fp32, one thread per (b,o).
// ---------------------------------------------------------------------------
__global__ __launch_bounds__(256) void kan_fallback_kernel(
    const float* __restrict__ x, const float* __restrict__ w,
    const float* __restrict__ bias, const float* __restrict__ centers,
    const float* __restrict__ widths, float* __restrict__ out)
{
  const int idx = blockIdx.x * 256 + threadIdx.x;
  const int b = idx / O_DIM, o = idx % O_DIM;
  float cc[K_RBF], iw[K_RBF];
#pragma unroll
  for (int k = 0; k < K_RBF; ++k) {
    cc[k] = centers[k];
    const float wv = widths[k];
    iw[k] = 0.5f / (wv * wv);
  }
  float s = 0.0f;
  for (int i = 0; i < I_DIM; ++i) {
    const float xv = x[(size_t)b * I_DIM + i];
    const float* wp = w + ((size_t)o * I_DIM + i) * K_RBF;
#pragma unroll
    for (int k = 0; k < K_RBF; ++k) {
      const float d = xv - cc[k];
      s = __builtin_fmaf(__expf(-d * d * iw[k]), wp[k], s);
    }
  }
  out[idx] = s + bias[o];
}

// ---------------------------------------------------------------------------
extern "C" void kernel_launch(void* const* d_in, const int* in_sizes, int n_in,
                              void* d_out, int out_size, void* d_ws, size_t ws_size,
                              hipStream_t stream) {
  (void)in_sizes; (void)n_in; (void)out_size;
  const float* x       = (const float*)d_in[0];
  const float* weights = (const float*)d_in[1];
  const float* bias    = (const float*)d_in[2];
  const float* centers = (const float*)d_in[3];
  const float* widths  = (const float*)d_in[4];
  float* out = (float*)d_out;

  const size_t actBytes = (size_t)B_DIM * KK * sizeof(_Float16);  // 128 MiB
  const size_t wBytes   = (size_t)O_DIM * KK * sizeof(_Float16);  //  32 MiB

  if (ws_size >= actBytes + wBytes) {
    _Float16* act = (_Float16*)d_ws;
    _Float16* wf  = (_Float16*)((char*)d_ws + actBytes);

    rbf_act_f16_kernel<<<(B_DIM * I_DIM) / 256, 256, 0, stream>>>(x, centers, widths, act);
    cvt_w_f16_kernel<<<(O_DIM * I_DIM) / 256, 256, 0, stream>>>(weights, wf);

    dim3 grid(O_DIM / BN, B_DIM / BM);   // (8, 32)
    kan_wmma_gemm_kernel<<<grid, 256, 0, stream>>>(act, wf, bias, out);
  } else {
    kan_fallback_kernel<<<(B_DIM * O_DIM) / 256, 256, 0, stream>>>(
        x, weights, bias, centers, widths, out);
  }
}
